// GATN_34291018891968
// MI455X (gfx1250) — matrix operand order repaired and checked by reading, not compiled, not run on hardware
//
#include <hip/hip_runtime.h>

// ---------------------------------------------------------------------------
// CDNA5 / gfx1250 implementation of the 3-layer GAT + annotation-MLP network.
//
// Dense matmuls: v_wmma_f32_16x16x32_f16, one wave per 16x16 tile, 4 waves per
// block. All A operands are zero-padded in workspace so the 32-wide K window
// of a fragment never needs bounds checks -> A loads are 4x global_load_b128.
// All B (weight) matrices are pre-packed to f16 in exact B-fragment order
// -> B loads are a single contiguous 32B read per lane. No divergence, EXEC
// stays all-ones (WMMA requirement).
//
// Edge softmax-aggregation: 3 memory-lean passes (atomic max, exp-accumulate,
// normalize) with recomputed logits; ~1.4 GB of L2-resident traffic total.
// ---------------------------------------------------------------------------

typedef __attribute__((ext_vector_type(16))) _Float16 v16h;
typedef __attribute__((ext_vector_type(8)))  float    v8f;

#define FDIM 15
#define LDH  16      // ld of BH / BA (edge-side buffers)
#define LDA  32      // ld of GEMM-A node buffers (zero-padded to K window)
#define NEG_SLOPE 0.2f

__device__ __forceinline__ float leakyrelu(float v) {
    return v > 0.f ? v : NEG_SLOPE * v;
}

// Float atomic-max via sign-split integer atomics (valid with -inf init).
__device__ __forceinline__ void atomicMaxF(float* addr, float val) {
    if (val >= 0.f) atomicMax((int*)addr, __float_as_int(val));
    else            atomicMin((unsigned int*)addr, __float_as_uint(val));
}

// ------------------------------ utility kernels ----------------------------

__global__ void zero_kernel(float* __restrict__ p, long n) {
    long i      = (long)blockIdx.x * blockDim.x + threadIdx.x;
    long stride = (long)gridDim.x * blockDim.x;
    for (; i < n; i += stride) p[i] = 0.f;
}

// dst[r*dld + c] = src[r*scols + c]   (padding pre-zeroed by zero_kernel)
__global__ void copy_pad_kernel(float* __restrict__ dst, const float* __restrict__ src,
                                long total, int scols, int dld) {
    long i = (long)blockIdx.x * blockDim.x + threadIdx.x;
    if (i >= total) return;
    long r = i / scols;
    int  c = (int)(i - r * scols);
    dst[r * (long)dld + c] = src[i];
}

// Pack f32 weight B[K,Nc] (row-major, ld=Nc) into f16 WMMA B-fragment order:
// Bh[(((tn*nkt)+kt)*32 + lane)*16 + e] = B[kt*32 + (lane<16?0:8)+(e<8?e:e+8), tn*16+(lane&15)]
// with zeros for out-of-range K / Nc.
__global__ void pack_b_kernel(const float* __restrict__ B, _Float16* __restrict__ Bh,
                              int K, int Nc, int total) {
    int idx = blockIdx.x * blockDim.x + threadIdx.x;
    if (idx >= total) return;
    int e    = idx & 15;
    int lane = (idx >> 4) & 31;
    int tkn  = idx >> 9;             // tn*nkt + kt  (nkt recovered below)
    int nkt  = (K + 31) >> 5;
    int kt   = tkn % nkt;
    int tn   = tkn / nkt;
    int col  = tn * 16 + (lane & 15);
    int k    = kt * 32 + ((lane < 16) ? 0 : 8) + ((e < 8) ? e : e + 8);
    int kc   = k   < K  ? k   : K  - 1;    // clamp -> unconditional load
    int cc   = col < Nc ? col : Nc - 1;
    float v  = B[(size_t)kc * Nc + cc];
    Bh[idx]  = (_Float16)((k < K && col < Nc) ? v : 0.f);
}

// ------------------------------ WMMA GEMM ----------------------------------
// C[Mpad, ldc] = act(A[Mpad, lda] @ Bh + bias). A rows/cols are padded so all
// loads are unconditional b128s. Grid: (Mpad/64, ntn), block (32,4).
__global__ void wmma_gemm_kernel(const float* __restrict__ A,
                                 const _Float16* __restrict__ Bh,
                                 const float* __restrict__ bias,
                                 float* __restrict__ C,
                                 int Nc, int lda, int ldc, int nkt, int act)
{
    const int  lane = threadIdx.x;                    // 0..31
    const int  tm   = blockIdx.x * blockDim.y + threadIdx.y;
    const int  tn   = blockIdx.y;
    const bool lo   = lane < 16;
    const int  l15  = lane & 15;
    const size_t aoff  = (size_t)(tm * 16 + l15) * lda + (lo ? 0 : 8);
    const _Float16* bp = Bh + ((size_t)tn * nkt * 32 + lane) * 16;

    v8f c = {};
    for (int kt = 0; kt < nkt; ++kt) {
        const float4* pa = (const float4*)(A + aoff + kt * 32);
        float4 a0 = pa[0];             // K run [base, base+8)
        float4 a1 = pa[1];
        float4 a2 = pa[4];             // K run [base+16, base+24)
        float4 a3 = pa[5];
        v16h a;
        a[0]  = (_Float16)a0.x; a[1]  = (_Float16)a0.y;
        a[2]  = (_Float16)a0.z; a[3]  = (_Float16)a0.w;
        a[4]  = (_Float16)a1.x; a[5]  = (_Float16)a1.y;
        a[6]  = (_Float16)a1.z; a[7]  = (_Float16)a1.w;
        a[8]  = (_Float16)a2.x; a[9]  = (_Float16)a2.y;
        a[10] = (_Float16)a2.z; a[11] = (_Float16)a2.w;
        a[12] = (_Float16)a3.x; a[13] = (_Float16)a3.y;
        a[14] = (_Float16)a3.z; a[15] = (_Float16)a3.w;

        v16h b = *(const v16h*)(bp + (size_t)kt * 512);   // 32B contiguous

        c = __builtin_amdgcn_wmma_f32_16x16x32_f16(false, a, false, b,
                                                   (short)0, c, false, false);
    }

    const int  bcol  = tn * 16 + l15;
    const bool colok = bcol < Nc;
    float bv = 0.f;
    if (bias != nullptr) bv = bias[colok ? bcol : (Nc - 1)];
#pragma unroll
    for (int r = 0; r < 8; ++r) {
        int   row = tm * 16 + r + (lo ? 0 : 8);
        float v   = colok ? (c[r] + bv) : 0.f;
        if (act == 1)      v = fmaxf(v, 0.f);
        else if (act == 2) v = 1.f / (1.f + __expf(-v));
        C[(size_t)row * ldc + bcol] = v;
    }
}

// ------------------------------ GAT kernels --------------------------------

__global__ void node_scores_kernel(const float* __restrict__ H,
                                   const float* __restrict__ as,
                                   const float* __restrict__ ad,
                                   float* __restrict__ ssrc,
                                   float* __restrict__ sdst, int n)
{
    int i = blockIdx.x * blockDim.x + threadIdx.x;
    if (i >= n) return;
    const float* h = H + (size_t)i * LDH;
    float s = 0.f, d = 0.f;
#pragma unroll
    for (int f = 0; f < FDIM; ++f) { s += h[f] * as[f]; d += h[f] * ad[f]; }
    ssrc[i] = s; sdst[i] = d;
}

__global__ void gat_init_kernel(float* __restrict__ mx, float* __restrict__ den,
                                float* __restrict__ acc, int n)
{
    int i = blockIdx.x * blockDim.x + threadIdx.x;
    if (i >= n) return;
    mx[i]  = -__builtin_inff();
    den[i] = 0.f;
    float4* a = (float4*)(acc + (size_t)i * LDH);
    float4 z  = {0.f, 0.f, 0.f, 0.f};
    a[0] = z; a[1] = z; a[2] = z; a[3] = z;
}

__global__ void edge_max_kernel(const int* __restrict__ ei, int E, int n,
                                const float* __restrict__ ssrc,
                                const float* __restrict__ sdst,
                                float* __restrict__ mx)
{
    long i   = (long)blockIdx.x * blockDim.x + threadIdx.x;
    long tot = (long)E + n;
    if (i >= tot) return;
    int s, d;
    if (i < E) { s = ei[i]; d = ei[(long)E + i]; }
    else       { s = d = (int)(i - E); }
    atomicMaxF(&mx[d], leakyrelu(ssrc[s] + sdst[d]));
}

__global__ void edge_acc_kernel(const int* __restrict__ ei, int E, int n,
                                const float* __restrict__ ssrc,
                                const float* __restrict__ sdst,
                                const float* __restrict__ mx,
                                const float* __restrict__ H,
                                float* __restrict__ den,
                                float* __restrict__ acc)
{
    long i   = (long)blockIdx.x * blockDim.x + threadIdx.x;
    long tot = (long)E + n;
    if (i >= tot) return;
    int s, d;
    if (i < E) { s = ei[i]; d = ei[(long)E + i]; }
    else       { s = d = (int)(i - E); }
    float e  = leakyrelu(ssrc[s] + sdst[d]);
    float ex = __expf(e - mx[d]);
    atomicAdd(&den[d], ex);
    const float4* hs = (const float4*)(H + (size_t)s * LDH);
    float4 h0 = hs[0], h1 = hs[1], h2 = hs[2], h3 = hs[3];
    float hv[16] = {h0.x, h0.y, h0.z, h0.w, h1.x, h1.y, h1.z, h1.w,
                    h2.x, h2.y, h2.z, h2.w, h3.x, h3.y, h3.z, h3.w};
    float* ac = acc + (size_t)d * LDH;
#pragma unroll
    for (int f = 0; f < FDIM; ++f) atomicAdd(&ac[f], ex * hv[f]);
}

// out(ld=LDA) = acc/(den+eps) + bias; cols 15..31 stay zero from init.
__global__ void gat_final_kernel(const float* __restrict__ acc,
                                 const float* __restrict__ den,
                                 const float* __restrict__ bias,
                                 float* __restrict__ out, int n)
{
    int i = blockIdx.x * blockDim.x + threadIdx.x;
    if (i >= n) return;
    float inv = 1.f / (den[i] + 1e-16f);
    const float* a = acc + (size_t)i * LDH;
    float*       o = out + (size_t)i * LDA;
#pragma unroll
    for (int f = 0; f < FDIM; ++f) o[f] = a[f] * inv + bias[f];
}

// ------------------------------ pooling / head -----------------------------

__global__ void pool_init_kernel(float* __restrict__ gsum, float* __restrict__ gmax,
                                 float* __restrict__ gcnt, int g)
{
    int i = blockIdx.x * blockDim.x + threadIdx.x;
    if (i >= g) return;
    gcnt[i] = 0.f;
#pragma unroll
    for (int f = 0; f < FDIM; ++f) {
        gsum[(size_t)i * LDH + f] = 0.f;
        gmax[(size_t)i * LDH + f] = -__builtin_inff();
    }
}

__global__ void pool_acc_kernel(const float* __restrict__ H,   // ld = LDA
                                const int* __restrict__ batch,
                                float* __restrict__ gsum, float* __restrict__ gmax,
                                float* __restrict__ gcnt, int n)
{
    int i = blockIdx.x * blockDim.x + threadIdx.x;
    if (i >= n) return;
    int g = batch[i];
    const float* h = H + (size_t)i * LDA;
    atomicAdd(&gcnt[g], 1.f);
#pragma unroll
    for (int f = 0; f < FDIM; ++f) {
        atomicAdd(&gsum[(size_t)g * LDH + f], h[f]);
        atomicMaxF(&gmax[(size_t)g * LDH + f], h[f]);
    }
}

// z = [sum(15) | max(15) | mean(15) | xann(45) | zeros(6)]  (ld = 96)
__global__ void build_z_kernel(const float* __restrict__ gsum,
                               const float* __restrict__ gmax,
                               const float* __restrict__ gcnt,
                               const float* __restrict__ xann,
                               float* __restrict__ Z, int G)
{
    int g = blockIdx.x * blockDim.x + threadIdx.x;
    if (g >= G) return;
    float  cnt = fmaxf(gcnt[g], 1.f);
    float* z   = Z + (size_t)g * 96;
#pragma unroll
    for (int f = 0; f < FDIM; ++f) {
        float s = gsum[(size_t)g * LDH + f];
        z[f]      = s;
        z[15 + f] = gmax[(size_t)g * LDH + f];
        z[30 + f] = s / cnt;
    }
#pragma unroll
    for (int j = 0; j < 45; ++j) z[45 + j] = xann[(size_t)g * 48 + j];
#pragma unroll
    for (int j = 90; j < 96; ++j) z[j] = 0.f;
}

__global__ void head_kernel(const float* __restrict__ Z1,
                            const float* __restrict__ w,
                            const float* __restrict__ b,
                            float* __restrict__ out, int G)
{
    int g = blockIdx.x * blockDim.x + threadIdx.x;
    if (g >= G) return;
    float s = b[0];
    const float* z = Z1 + (size_t)g * 96;
#pragma unroll
    for (int j = 0; j < 90; ++j) s += z[j] * w[j];
    out[g] = 1.f / (1.f + __expf(-s));
}

// ---------------------------------------------------------------------------
extern "C" void kernel_launch(void* const* d_in, const int* in_sizes, int n_in,
                              void* d_out, int out_size, void* d_ws, size_t ws_size,
                              hipStream_t stream)
{
    const float* x     = (const float*)d_in[0];
    const int*   ei    = (const int*)d_in[1];
    const int*   batch = (const int*)d_in[2];
    const float* xA    = (const float*)d_in[3];

    const float* cW[3]  = {(const float*)d_in[4],  (const float*)d_in[10], (const float*)d_in[16]};
    const float* cas[3] = {(const float*)d_in[5],  (const float*)d_in[11], (const float*)d_in[17]};
    const float* cad[3] = {(const float*)d_in[6],  (const float*)d_in[12], (const float*)d_in[18]};
    const float* cb[3]  = {(const float*)d_in[7],  (const float*)d_in[13], (const float*)d_in[19]};
    const float* fW[3]  = {(const float*)d_in[8],  (const float*)d_in[14], (const float*)d_in[20]};
    const float* fb[3]  = {(const float*)d_in[9],  (const float*)d_in[15], (const float*)d_in[21]};
    const float* m1W = (const float*)d_in[22];
    const float* m1b = (const float*)d_in[23];
    const float* aW1 = (const float*)d_in[24];
    const float* ab1 = (const float*)d_in[25];
    const float* aW2 = (const float*)d_in[26];
    const float* ab2 = (const float*)d_in[27];
    const float* aW3 = (const float*)d_in[28];
    const float* ab3 = (const float*)d_in[29];
    const float* aW4 = (const float*)d_in[30];
    const float* ab4 = (const float*)d_in[31];
    const float* m2W = (const float*)d_in[32];
    const float* m2b = (const float*)d_in[33];

    const int N    = in_sizes[0] / FDIM;
    const int E    = in_sizes[1] / 2;
    const int G    = in_sizes[3] / 300;
    const int Npad = (N + 63) & ~63;

    // ------------------- workspace layout (float units, 32B aligned) -------
    float* ws = (float*)d_ws;
    size_t o = 0;
    auto alloc = [&](size_t nf) { float* p = ws + o; o += (nf + 7) & ~(size_t)7; return p; };

    // zero-initialized region (padding must be zero) -------------------------
    float* XP  = alloc((size_t)Npad * LDA);   // padded x           [Npad,32]
    float* BO  = alloc((size_t)Npad * LDA);   // GAT output         [Npad,32]
    float* BF  = alloc((size_t)Npad * LDA);   // FC output          [Npad,32]
    float* XAP = alloc((size_t)G * 320);      // padded xA          [G,320]
    float* A1  = alloc((size_t)G * 320);
    float* A2  = alloc((size_t)G * 320);
    const long zeroN = (long)(o);             // everything above gets zeroed
    // fully-overwritten region ----------------------------------------------
    float* BH   = alloc((size_t)Npad * LDH);  // h = x @ W          [Npad,16]
    float* BA   = alloc((size_t)Npad * LDH);  // softmax accumulator
    float* ssrc = alloc((size_t)N);
    float* sdst = alloc((size_t)N);
    float* mx   = alloc((size_t)N);
    float* den  = alloc((size_t)N);
    float* gsum = alloc((size_t)G * LDH);
    float* gmax = alloc((size_t)G * LDH);
    float* gcnt = alloc((size_t)G);
    float* XANN = alloc((size_t)G * 48);
    float* Z    = alloc((size_t)G * 96);
    float* Z1   = alloc((size_t)G * 96);
    // packed f16 weights (sizes in halfs = ntn*nkt*512) ---------------------
    auto allocH = [&](size_t nh) { _Float16* p = (_Float16*)(ws + o); o += (nh / 2 + 7) & ~(size_t)7; return p; };
    _Float16* cWp[3]; _Float16* fWp[3];
    for (int l = 0; l < 3; ++l) { cWp[l] = allocH(512); fWp[l] = allocH(512); }
    _Float16* aW1p = allocH((size_t)19 * 10 * 512);
    _Float16* aW2p = allocH((size_t)19 * 10 * 512);
    _Float16* aW3p = allocH((size_t)19 * 10 * 512);
    _Float16* aW4p = allocH((size_t)3  * 10 * 512);
    _Float16* m1Wp = allocH((size_t)6  * 3  * 512);
    (void)ws_size; (void)n_in; (void)out_size;

    const int TPB   = 256;
    const int nodeB = (N + TPB - 1) / TPB;
    const long tot  = (long)E + N;
    const int edgeB = (int)((tot + TPB - 1) / TPB);
    const int gB    = (G + TPB - 1) / TPB;

    auto gemm = [&](const float* A, const _Float16* Bh, const float* bias, float* C,
                    int Mpad, int Nc, int lda, int ldc, int nkt, int act) {
        dim3 grid(Mpad / 64, (Nc + 15) / 16);
        dim3 block(32, 4);
        hipLaunchKernelGGL(wmma_gemm_kernel, grid, block, 0, stream,
                           A, Bh, bias, C, Nc, lda, ldc, nkt, act);
    };
    auto pack = [&](const float* B, _Float16* Bh, int K, int Nc) {
        int total = ((Nc + 15) / 16) * ((K + 31) / 32) * 512;
        hipLaunchKernelGGL(pack_b_kernel, dim3((total + TPB - 1) / TPB), dim3(TPB),
                           0, stream, B, Bh, K, Nc, total);
    };

    // 1) zero padding region, stage padded inputs, pack weights
    hipLaunchKernelGGL(zero_kernel, dim3(2048), dim3(TPB), 0, stream, ws, zeroN);
    hipLaunchKernelGGL(copy_pad_kernel, dim3((int)(((long)N * FDIM + TPB - 1) / TPB)),
                       dim3(TPB), 0, stream, XP, x, (long)N * FDIM, FDIM, LDA);
    hipLaunchKernelGGL(copy_pad_kernel, dim3((int)(((long)G * 300 + TPB - 1) / TPB)),
                       dim3(TPB), 0, stream, XAP, xA, (long)G * 300, 300, 320);
    for (int l = 0; l < 3; ++l) { pack(cW[l], cWp[l], FDIM, FDIM); pack(fW[l], fWp[l], FDIM, FDIM); }
    pack(aW1, aW1p, 300, 300); pack(aW2, aW2p, 300, 300); pack(aW3, aW3p, 300, 300);
    pack(aW4, aW4p, 300, 45);  pack(m1W, m1Wp, 90, 90);

    // 2) annotation MLP on xA
    gemm(XAP, aW1p, ab1, A1,   G, 300, 320, 320, 10, 1);   // relu
    gemm(A1,  aW2p, ab2, A2,   G, 300, 320, 320, 10, 1);   // relu
    gemm(A2,  aW3p, ab3, A1,   G, 300, 320, 320, 10, 2);   // sigmoid
    gemm(A1,  aW4p, ab4, XANN, G, 45,  320, 48,  10, 0);   // linear

    // 3) GAT stack
    const float* X = XP;
    for (int l = 0; l < 3; ++l) {
        gemm(X, cWp[l], nullptr, BH, Npad, FDIM, LDA, LDH, 1, 0);
        hipLaunchKernelGGL(node_scores_kernel, dim3(nodeB), dim3(TPB), 0, stream,
                           BH, cas[l], cad[l], ssrc, sdst, N);
        hipLaunchKernelGGL(gat_init_kernel, dim3(nodeB), dim3(TPB), 0, stream,
                           mx, den, BA, N);
        hipLaunchKernelGGL(edge_max_kernel, dim3(edgeB), dim3(TPB), 0, stream,
                           ei, E, N, ssrc, sdst, mx);
        hipLaunchKernelGGL(edge_acc_kernel, dim3(edgeB), dim3(TPB), 0, stream,
                           ei, E, N, ssrc, sdst, mx, BH, den, BA);
        hipLaunchKernelGGL(gat_final_kernel, dim3(nodeB), dim3(TPB), 0, stream,
                           BA, den, cb[l], BO, N);
        gemm(BO, fWp[l], fb[l], BF, Npad, FDIM, LDA, LDA, 1, (l < 2) ? 1 : 0);
        X = BF;
    }

    // 4) pooling + head
    hipLaunchKernelGGL(pool_init_kernel, dim3(gB), dim3(TPB), 0, stream,
                       gsum, gmax, gcnt, G);
    hipLaunchKernelGGL(pool_acc_kernel, dim3(nodeB), dim3(TPB), 0, stream,
                       BF, batch, gsum, gmax, gcnt, N);
    hipLaunchKernelGGL(build_z_kernel, dim3(gB), dim3(TPB), 0, stream,
                       gsum, gmax, gcnt, XANN, Z, G);
    gemm(Z, m1Wp, m1b, Z1, G, 90, 96, 96, 3, 1);           // relu
    hipLaunchKernelGGL(head_kernel, dim3(gB), dim3(TPB), 0, stream,
                       Z1, m2W, m2b, (float*)d_out, G);
}
